// MixedMultiHeadAttention_23545010716838
// MI455X (gfx1250) — matrix operand order repaired and checked by reading, not compiled, not run on hardware
//
#include <hip/hip_runtime.h>
#include <hip/hip_bf16.h>
#include <limits.h>

// MI455X (gfx1250) mixed-mask multi-head attention.
// B=4, T=1024, F=1024, G=4, M=4 (16 heads), K=V=64, LOCAL_SCOPE=128.
// Compute-bound (~5.2e10 FLOPs vs ~4us of HBM traffic at 23.3 TB/s) ->
// everything on v_wmma_f32_16x16x32_bf16, f32 accumulation, f32 softmax.

typedef __attribute__((ext_vector_type(16))) __bf16   v16bf;
typedef __attribute__((ext_vector_type(8)))  float    v8f;
typedef __attribute__((ext_vector_type(4)))  unsigned uv4;
typedef __attribute__((ext_vector_type(4)))  float    fv4;
typedef __attribute__((ext_vector_type(2)))  unsigned uv2;

union Frag {
  v16bf    v;
  unsigned u[8];
  uv4      q[2];
};

static __device__ __forceinline__ unsigned short f2bf(float f) {
  union { float f; unsigned u; } x; x.f = f;
  unsigned u = x.u;
  u += 0x7FFFu + ((u >> 16) & 1u);   // round-to-nearest-even
  return (unsigned short)(u >> 16);
}
static __device__ __forceinline__ unsigned pack2(float a, float b) {
  return (unsigned)f2bf(a) | ((unsigned)f2bf(b) << 16);
}

// ---------------------------------------------------------------------------
// f32 -> bf16 bulk convert (4 elements / thread)
// ---------------------------------------------------------------------------
__global__ __launch_bounds__(256)
void cvt_f32_bf16(const float* __restrict__ src, unsigned short* __restrict__ dst, int n4) {
  int i = blockIdx.x * 256 + threadIdx.x;
  if (i < n4) {
    fv4 f = ((const fv4*)src)[i];
    uv2 o;
    o.x = pack2(f.x, f.y);
    o.y = pack2(f.z, f.w);
    ((uv2*)dst)[i] = o;
  }
}

// ---------------------------------------------------------------------------
// Generic bf16 WMMA GEMM:  C[row, col] = (sum_k A[row,k]*W[k,col] + bias[col]) * alpha
//   A: bf16 [Mrows x 1024] row-major (lda = 1024)
//   W: f32  [1024 x ldw]   row-major, per-head stride wHeadStride
//   CMODE 0: bf16 row-major [row*ldc + col]      (Q, K storage)
//   CMODE 1: bf16 transposed [col*ldc + row]     (V stored as Vt[v][t])
//   CMODE 2: f32  row-major                      (final output)
// Block tile 128x64, 8 waves; wave w -> rows (w/2)*32, cols (w&1)*32 (2x2 tiles).
// Double-buffered LDS: next tile's global loads overlap current tile's WMMAs.
// ---------------------------------------------------------------------------
#define AP 20   // A LDS pitch in uints (80B, 16B aligned, bank-spread)
#define BP 20   // B LDS pitch in uints

template <int CMODE>
__global__ __launch_bounds__(256)
void gemm_bf16_wmma(const unsigned short* __restrict__ A,
                    const float* __restrict__ W, int ldw,
                    const float* __restrict__ bias, float alpha,
                    void* __restrict__ Cout, int ldc,
                    long wHeadStride, long cHeadStride)
{
  __shared__ unsigned As[2][128 * AP];
  __shared__ unsigned Bs[2][64 * BP];   // stored transposed: Bs[n][k] bf16

  const int tid  = threadIdx.x;
  const int wave = tid >> 5, lane = tid & 31;
  const int lrow = lane & 15, hi = lane >> 4;
  const int h    = blockIdx.z;
  const long bm  = (long)blockIdx.y * 128;
  const int  bn  = blockIdx.x * 64;
  const float* Wh = W + (long)h * wHeadStride;
  const int wr = (wave >> 1) * 32;
  const int wc = (wave & 1) * 32;

  // per-thread tile-staging indices (A tile 128x32 bf16, W tile 32x64 f32)
  const int arow = tid >> 4,  ac = tid & 15;   // + it*16 rows
  const int wkr  = tid >> 6,  wn = tid & 63;   // + it*4  k-rows

  unsigned aReg[8];
  float    wReg[8];

  v8f acc[2][2] = {};

  // ---- prologue: tile 0 -> buffer 0 ----
  #pragma unroll
  for (int it = 0; it < 8; ++it)
    aReg[it] = *(const unsigned*)(A + (bm + arow + it * 16) * 1024 + ac * 2);
  #pragma unroll
  for (int it = 0; it < 8; ++it)
    wReg[it] = Wh[(long)(wkr + it * 4) * ldw + bn + wn];
  #pragma unroll
  for (int it = 0; it < 8; ++it)
    As[0][(arow + it * 16) * AP + ac] = aReg[it];
  #pragma unroll
  for (int it = 0; it < 8; ++it)
    ((unsigned short*)Bs[0])[wn * (BP * 2) + wkr + it * 4] = f2bf(wReg[it]);
  __syncthreads();

  int cur = 0;
  for (int kk = 0; kk < 1024; kk += 32) {
    const bool more = (kk + 32) < 1024;
    // issue next tile's global loads (latency hidden under WMMAs below)
    if (more) {
      #pragma unroll
      for (int it = 0; it < 8; ++it)
        aReg[it] = *(const unsigned*)(A + (bm + arow + it * 16) * 1024 + (kk + 32) + ac * 2);
      #pragma unroll
      for (int it = 0; it < 8; ++it)
        wReg[it] = Wh[(long)(kk + 32 + wkr + it * 4) * ldw + bn + wn];
    }

    Frag a[2], b[2];
    // A frag (16x32): lanes 0-15 K{0-7,16-23}, lanes 16-31 K{8-15,24-31}
    #pragma unroll
    for (int i = 0; i < 2; ++i) {
      const unsigned* base = &As[cur][(wr + i * 16 + lrow) * AP];
      a[i].q[0] = *(const uv4*)(base + hi * 4);
      a[i].q[1] = *(const uv4*)(base + 8 + hi * 4);
    }
    // B frag (32x16): lane n = column; lanes 0-15 K{0-7,16-23}, lanes 16-31 K{8-15,24-31}
    #pragma unroll
    for (int j = 0; j < 2; ++j) {
      const unsigned* base = &Bs[cur][(wc + j * 16 + lrow) * BP];
      b[j].q[0] = *(const uv4*)(base + hi * 4);
      b[j].q[1] = *(const uv4*)(base + 8 + hi * 4);
    }
    #pragma unroll
    for (int i = 0; i < 2; ++i)
      #pragma unroll
      for (int j = 0; j < 2; ++j)
        acc[i][j] = __builtin_amdgcn_wmma_f32_16x16x32_bf16(
            false, a[i].v, false, b[j].v, (short)0, acc[i][j], false, false);

    if (more) {
      const int nxt = cur ^ 1;
      #pragma unroll
      for (int it = 0; it < 8; ++it)
        As[nxt][(arow + it * 16) * AP + ac] = aReg[it];
      #pragma unroll
      for (int it = 0; it < 8; ++it)
        ((unsigned short*)Bs[nxt])[wn * (BP * 2) + wkr + it * 4] = f2bf(wReg[it]);
    }
    __syncthreads();
    cur ^= 1;
  }

  // epilogue: C/D layout VGPR r -> row r + (lane<16?0:8), col = lane&15
  #pragma unroll
  for (int i = 0; i < 2; ++i) {
    #pragma unroll
    for (int j = 0; j < 2; ++j) {
      int cl   = wc + j * 16 + lrow;
      int colg = bn + cl;
      float bsv = bias[colg];
      #pragma unroll
      for (int r = 0; r < 8; ++r) {
        long row = bm + wr + i * 16 + hi * 8 + r;
        float v = (acc[i][j][r] + bsv) * alpha;
        if (CMODE == 0)
          ((unsigned short*)Cout + (long)h * cHeadStride)[row * ldc + colg] = f2bf(v);
        else if (CMODE == 1)
          ((unsigned short*)Cout + (long)h * cHeadStride)[(long)colg * ldc + row] = f2bf(v);
        else
          ((float*)Cout)[row * ldc + colg] = v;
      }
    }
  }
}

// ---------------------------------------------------------------------------
// Attention: one block per (b, head, 64 query rows). Full 64x1024 score strip
// kept in LDS (f32, pitch 1028 = 4*257: rows 16B-aligned for b128 LDS ops,
// odd quotient spreads rows over all 16 b128 bank groups). 320KB/WGP makes
// full materialization viable -- no online softmax needed. Masks are
// branch-free: group -> uniform [lo, hi] bounds on d = i - j.
// ---------------------------------------------------------------------------
#define SP 1028          // score LDS pitch (dwords, 16B-aligned rows)
#define QP 36            // Q LDS pitch (uints, 144B)
#define NEGM (-1e30f)

__global__ __launch_bounds__(256)
void attn_wmma(const unsigned short* __restrict__ Q,
               const unsigned short* __restrict__ K,
               const unsigned short* __restrict__ Vt,
               unsigned short* __restrict__ Out)
{
  extern __shared__ unsigned char smem[];
  float*    S   = (float*)smem;                                 // 64 x SP f32
  unsigned* Qs  = (unsigned*)(smem + (size_t)64 * SP * 4);      // 64 x QP uints
  float*    red = (float*)(smem + (size_t)64 * SP * 4 + 64 * QP * 4); // 512 f32

  const int tid  = threadIdx.x;
  const int wave = tid >> 5, lane = tid & 31;
  const int lrow = lane & 15, hi = lane >> 4;
  const int b = blockIdx.z, h = blockIdx.y, q0 = blockIdx.x * 64;
  const int g = h >> 2;                       // mask group
  // branch-free mask bounds on d = (query idx) - (key idx):
  //   g0 global: (INT_MIN, INT_MAX)   g1 local: [-128, 128]
  //   g2 fwd (mask j<i, d>0): (INT_MIN, 0]   g3 bwd (mask j>i, d<0): [0, INT_MAX)
  const int lo = (g == 1) ? -128 : ((g == 3) ? 0 : INT_MIN);
  const int hb = (g == 1) ?  128 : ((g == 2) ? 0 : INT_MAX);
  const long headOff = (long)h * (4096 * 64);
  const unsigned short* Qh = Q  + headOff + (long)(b * 1024 + q0) * 64;
  const unsigned short* Kh = K  + headOff + (long)(b * 1024) * 64;
  const unsigned short* Vh = Vt + headOff + (long)(b * 1024);   // Vt[v][t], v-stride 4096

  // stage Q tile (64x64 bf16, already scaled by 1/sqrt(K) in projection)
  #pragma unroll
  for (int it = 0; it < 8; ++it) {
    int idx = tid + it * 256;
    int row = idx >> 5, c = idx & 31;
    Qs[row * QP + c] = *(const unsigned*)(Qh + row * 64 + c * 2);
  }
  __syncthreads();

  // ---- scores: wave w -> row block w/2 (16 rows), col blocks (w&1)+2n ----
  const int rb = wave >> 1;
  Frag aQ[2];
  #pragma unroll
  for (int ks = 0; ks < 2; ++ks) {
    const unsigned* base = &Qs[(rb * 16 + lrow) * QP + ks * 16];
    aQ[ks].q[0] = *(const uv4*)(base + hi * 4);
    aQ[ks].q[1] = *(const uv4*)(base + 8 + hi * 4);
  }
  #pragma unroll 2
  for (int cb = (wave & 1); cb < 64; cb += 2) {
    v8f acc = {};
    #pragma unroll
    for (int ks = 0; ks < 2; ++ks) {
      Frag bK;     // B = K^T: lane n reads K-storage row (key t), contiguous
      const unsigned short* kp = Kh + (long)(cb * 16 + lrow) * 64 + ks * 32 + hi * 8;
      bK.q[0] = *(const uv4*)kp;
      bK.q[1] = *(const uv4*)(kp + 16);
      acc = __builtin_amdgcn_wmma_f32_16x16x32_bf16(
          false, aQ[ks].v, false, bK.v, (short)0, acc, false, false);
    }
    const int j  = cb * 16 + lrow;             // key index
    const int d0 = (q0 + rb * 16 + hi * 8) - j;
    float* so = &S[(rb * 16 + hi * 8) * SP + j];
    #pragma unroll
    for (int r = 0; r < 8; ++r) {
      int d = d0 + r;                          // i - j as in reference
      bool masked = (d < lo) | (d > hb);
      so[r * SP] = masked ? NEGM : acc[r];
    }
  }
  __syncthreads();

  // ---- softmax: 4 threads/row, blocked 256-col chunks, b128 LDS traffic;
  //      keep exp unnormalized, fold 1/rowsum into the P@V epilogue.
  {
    int row = tid & 63, chunk = tid >> 6;
    float* sr = &S[row * SP + chunk * 256];
    float mx = -3.4e38f;
    for (int s = 0; s < 256; s += 4) {
      fv4 x = *(const fv4*)(sr + s);
      mx = fmaxf(fmaxf(fmaxf(mx, x.x), x.y), fmaxf(x.z, x.w));
    }
    red[row * 4 + chunk] = mx;
    __syncthreads();
    float rmax = fmaxf(fmaxf(red[row * 4 + 0], red[row * 4 + 1]),
                       fmaxf(red[row * 4 + 2], red[row * 4 + 3]));
    float sum = 0.0f;
    for (int s = 0; s < 256; s += 4) {
      fv4 x = *(const fv4*)(sr + s);
      x.x = __expf(x.x - rmax);
      x.y = __expf(x.y - rmax);
      x.z = __expf(x.z - rmax);
      x.w = __expf(x.w - rmax);
      *(fv4*)(sr + s) = x;
      sum += (x.x + x.y) + (x.z + x.w);
    }
    red[256 + row * 4 + chunk] = sum;
  }
  __syncthreads();

  // ---- O = P @ V, divide by rowsum at the end ----
  v8f oacc[2] = {};
  const int rb2 = wave >> 1;
  const int cbs = (wave & 1) * 2;              // 2 of 4 v-dim col blocks
  for (int s0 = 0; s0 < 1024; s0 += 32) {
    Frag aP;                                    // pack exp'd f32 scores -> bf16 pairs
    const float* sp1 = &S[(rb2 * 16 + lrow) * SP + s0 + hi * 8];
    fv4 x0 = *(const fv4*)(sp1);
    fv4 x1 = *(const fv4*)(sp1 + 4);
    fv4 x2 = *(const fv4*)(sp1 + 16);
    fv4 x3 = *(const fv4*)(sp1 + 20);
    aP.u[0] = pack2(x0.x, x0.y);  aP.u[1] = pack2(x0.z, x0.w);
    aP.u[2] = pack2(x1.x, x1.y);  aP.u[3] = pack2(x1.z, x1.w);
    aP.u[4] = pack2(x2.x, x2.y);  aP.u[5] = pack2(x2.z, x2.w);
    aP.u[6] = pack2(x3.x, x3.y);  aP.u[7] = pack2(x3.z, x3.w);
    #pragma unroll
    for (int jj = 0; jj < 2; ++jj) {
      Frag bV;   // B = V: lane n reads Vt row (v index), contiguous in t
      const unsigned short* vp = Vh + (long)((cbs + jj) * 16 + lrow) * 4096 + s0 + hi * 8;
      bV.q[0] = *(const uv4*)vp;
      bV.q[1] = *(const uv4*)(vp + 16);
      oacc[jj] = __builtin_amdgcn_wmma_f32_16x16x32_bf16(
          false, aP.v, false, bV.v, (short)0, oacc[jj], false, false);
    }
  }
  #pragma unroll
  for (int jj = 0; jj < 2; ++jj) {
    int ch = h * 64 + (cbs + jj) * 16 + lrow;  // channel = ((g*4+m)*64 + v)
    #pragma unroll
    for (int r = 0; r < 8; ++r) {
      int i = rb2 * 16 + hi * 8 + r;
      float rs = red[256 + i * 4 + 0] + red[256 + i * 4 + 1]
               + red[256 + i * 4 + 2] + red[256 + i * 4 + 3];
      float v = oacc[jj][r] / rs;
      Out[(long)(b * 1024 + q0 + i) * 1024 + ch] = f2bf(v);
    }
  }
}

// ---------------------------------------------------------------------------
extern "C" void kernel_launch(void* const* d_in, const int* in_sizes, int n_in,
                              void* d_out, int out_size, void* d_ws, size_t ws_size,
                              hipStream_t stream)
{
  const float* query = (const float*)d_in[0];
  const float* value = (const float*)d_in[1];
  const float* Wq    = (const float*)d_in[2];
  const float* bq    = (const float*)d_in[3];
  const float* Wk    = (const float*)d_in[4];
  const float* bk    = (const float*)d_in[5];
  const float* Wv    = (const float*)d_in[6];
  const float* bv    = (const float*)d_in[7];
  const float* Wo    = (const float*)d_in[8];
  const float* bo    = (const float*)d_in[9];

  unsigned char* ws = (unsigned char*)d_ws;
  const long SEG = (long)4096 * 1024 * 2;              // 8 MiB bf16 [4096 x 1024]
  unsigned short* xq     = (unsigned short*)(ws);            // query bf16
  unsigned short* xv     = (unsigned short*)(ws + SEG);      // value bf16
  unsigned short* q_ws   = (unsigned short*)(ws + 2 * SEG);  // [16][4096][64]
  unsigned short* k_ws   = (unsigned short*)(ws + 3 * SEG);  // [16][4096][64]
  unsigned short* vt_ws  = (unsigned short*)(ws + 4 * SEG);  // [16][64][4096] (transposed)
  unsigned short* att_ws = (unsigned short*)(ws + 5 * SEG);  // [4096][1024] btgmv

  const int n4 = 4096 * 1024 / 4;
  cvt_f32_bf16<<<n4 / 256, 256, 0, stream>>>(query, xq, n4);
  cvt_f32_bf16<<<n4 / 256, 256, 0, stream>>>(value, xv, n4);

  // projections: per-head [4096x1024]x[1024x64], 16 heads on grid.z
  dim3 gp(1, 32, 16);
  gemm_bf16_wmma<0><<<gp, 256, 0, stream>>>(xq, Wq, 64, bq, 0.125f, q_ws, 64,
                                            (long)1024 * 64, (long)4096 * 64);
  gemm_bf16_wmma<0><<<gp, 256, 0, stream>>>(xv, Wk, 64, bk, 1.0f, k_ws, 64,
                                            (long)1024 * 64, (long)4096 * 64);
  gemm_bf16_wmma<1><<<gp, 256, 0, stream>>>(xv, Wv, 64, bv, 1.0f, vt_ws, 4096,
                                            (long)1024 * 64, (long)4096 * 64);

  // attention: (rowTile=16, head=16, batch=4); ~275KB dynamic LDS (<320KB/WGP)
  size_t smem = (size_t)64 * SP * 4 + (size_t)64 * QP * 4 + 512 * 4;
  dim3 ga(16, 16, 4);
  attn_wmma<<<ga, 256, smem, stream>>>(q_ws, k_ws, vt_ws, att_ws);

  // output projection: [4096x1024]x[1024x1024] + bo -> f32 d_out
  dim3 go(16, 32, 1);
  gemm_bf16_wmma<2><<<go, 256, 0, stream>>>(att_ws, Wo, 1024, bo, 1.0f, (float*)d_out,
                                            1024, 0, 0);
}
